// CustomizedMultiHeadAttention_4063039062623
// MI455X (gfx1250) — matrix operand order repaired and checked by reading, not compiled, not run on hardware
//
#include <hip/hip_runtime.h>
#include <hip/hip_bf16.h>

// ---------------------------------------------------------------------------
// CustomizedMultiHeadAttention on MI455X (gfx1250), bf16 WMMA pipeline.
// B=4, S=2048, D=1024, H=16, DH=64.
// d_out = [ out (B*S*D f32) | attn_weights (B*H*S*S f32) ]
// Round 4: fix nontemporal builtin by using native clang ext-vector f32x2.
// Half-offset folded into base pointers -> all fragment loads and epilogue
// accesses use compile-time immediate offsets; non-temporal hints on the
// 1GB attn stream; 32x64 block per wave (8 WMMA / k-step).
// ---------------------------------------------------------------------------

#define B_ 4
#define S_ 2048
#define D_ 1024
#define H_ 16
#define DH_ 64
#define OUT_ELEMS ((size_t)B_ * S_ * D_)          // 8,388,608
#define ATTN_ELEMS ((size_t)B_ * H_ * S_ * S_)    // 268,435,456

typedef __attribute__((ext_vector_type(16))) __bf16 v16bf;
typedef __attribute__((ext_vector_type(8)))  float  v8f;
typedef __attribute__((ext_vector_type(2)))  float  f32x2;

struct Frag { unsigned int u[8]; };   // 8 dwords = 16 bf16 = one WMMA operand

__device__ __forceinline__ unsigned short f32_to_bf16(float f) {
    unsigned int u = __float_as_uint(f);
    unsigned int r = u + 0x7FFFu + ((u >> 16) & 1u);   // round to nearest even
    return (unsigned short)(r >> 16);
}
__device__ __forceinline__ unsigned int pack_bf16x2(float lo, float hi) {
    return (unsigned int)f32_to_bf16(lo) | ((unsigned int)f32_to_bf16(hi) << 16);
}
__device__ __forceinline__ v8f wmma_bf16(const Frag& a, const Frag& b, v8f c) {
    return __builtin_amdgcn_wmma_f32_16x16x32_bf16(
        false, __builtin_bit_cast(v16bf, a),
        false, __builtin_bit_cast(v16bf, b),
        (short)0, c, false, false);
}

// --- Fragment loaders. All pointers are PRE-OFFSET by the lane-half term
// (A: +half*8 elements, B: +half*8 elements), so every per-j offset below
// is a compile-time constant -> immediate-offset vector loads.

// A fragment, bf16 source. p = row + k0 + half*8.
__device__ __forceinline__ Frag load_a_bf16(const unsigned short* __restrict__ p) {
    Frag f;
#pragma unroll
    for (int j = 0; j < 8; ++j) {
        const int off = ((j >> 2) << 4) + ((j & 3) << 1);   // 0,2,4,6,16,18,20,22
        f.u[j] = *reinterpret_cast<const unsigned int*>(p + off);
    }
    return f;
}
// A fragment, fp32 source converted to bf16. p = row + k0 + half*8.
template <bool NT>
__device__ __forceinline__ Frag load_a_f32(const float* __restrict__ p) {
    Frag f;
#pragma unroll
    for (int j = 0; j < 8; ++j) {
        const int off = ((j >> 2) << 4) + ((j & 3) << 1);
        const f32x2* q = reinterpret_cast<const f32x2*>(p + off);
        f32x2 v = NT ? __builtin_nontemporal_load(q) : *q;
        f.u[j] = pack_bf16x2(v.x, v.y);
    }
    return f;
}
// B fragment, bf16 source. p = row + k0 + half*8.
__device__ __forceinline__ Frag load_b_bf16(const unsigned short* __restrict__ p) {
    Frag f;
#pragma unroll
    for (int j = 0; j < 8; ++j)
        f.u[j] = *reinterpret_cast<const unsigned int*>(p + ((j >> 2) << 4) + ((j & 3) << 1));
    return f;
}

// ---------------------------------------------------------------------------
// Kernel 1: transpose+convert weight fp32 [K=1024][N=1024] -> bf16 [N][K]
// ---------------------------------------------------------------------------
__global__ __launch_bounds__(256) void transpose_w_bf16(
    const float* __restrict__ W, unsigned short* __restrict__ Wt) {
    int idx = blockIdx.x * 256 + threadIdx.x;      // 1M elements
    int k = idx >> 10, n = idx & 1023;
    Wt[(size_t)n * 1024 + k] = f32_to_bf16(W[(size_t)k * 1024 + n]);
}

// ---------------------------------------------------------------------------
// Kernel 2: projection GEMM  P = X(8192x1024,f32) @ Wt(bf16 NxK) + bias.
// One wave computes a 32x64 block: 2 A-frags x 4 B-frags -> 8 WMMA / k-step.
// MODE 0: store bf16 [B,H,S,DH]   (Q, K projections)
// MODE 1: store bf16 [B,H,DH,S]   (V projection, transposed for attn@V)
// ---------------------------------------------------------------------------
template <int MODE>
__global__ __launch_bounds__(128) void proj_gemm(
    const float* __restrict__ X, const unsigned short* __restrict__ Wt,
    const float* __restrict__ bias, unsigned short* __restrict__ out) {
    const int wave = blockIdx.x * 4 + (threadIdx.x >> 5);  // 4096 waves
    const int mt = wave >> 4;          // 256 row blocks of 32
    const int ns = wave & 15;          // 16 col strips of 64
    const int lane = threadIdx.x & 31;
    const int half = lane >> 4;
    const int r = lane & 15;

    const float* __restrict__ pa0 =
        X + (size_t)(mt * 32 + r) * 1024 + (half << 3);
    const float* __restrict__ pa1 = pa0 + 16 * 1024;
    const unsigned short* __restrict__ pb[4];
#pragma unroll
    for (int t = 0; t < 4; ++t)
        pb[t] = Wt + (size_t)(ns * 64 + t * 16 + r) * 1024 + (half << 3);

    v8f acc[8] = {};                   // acc[mi*4 + t]
    for (int k0 = 0; k0 < 1024; k0 += 32) {
        __builtin_prefetch((const void*)(pa0 + 128), 0, 1);
        __builtin_prefetch((const void*)(pb[0] + 128), 0, 1);
        Frag fa0 = load_a_f32<false>(pa0);
        Frag fa1 = load_a_f32<false>(pa1);
        Frag fb[4];
#pragma unroll
        for (int t = 0; t < 4; ++t) fb[t] = load_b_bf16(pb[t]);
#pragma unroll
        for (int t = 0; t < 4; ++t) {
            acc[t]     = wmma_bf16(fa0, fb[t], acc[t]);
            acc[4 + t] = wmma_bf16(fa1, fb[t], acc[4 + t]);
        }
        pa0 += 32; pa1 += 32;
#pragma unroll
        for (int t = 0; t < 4; ++t) pb[t] += 32;
    }

    // Epilogue: 32 | 2048 so the whole block lies in one batch b.
    const int rowBase = mt * 32;                  // 0..8160
    const int b = rowBase >> 11, s0 = (rowBase & 2047) + (half << 3);
#pragma unroll
    for (int t = 0; t < 4; ++t) {
        const int c = ns * 64 + t * 16 + r;       // output column 0..1023
        const int h = c >> 6, dh = c & 63;
        const float bv = bias[c];
        unsigned short* __restrict__ po = (MODE == 0)
            ? out + (((size_t)b * H_ + h) * S_ + s0) * DH_ + dh
            : out + (((size_t)b * H_ + h) * DH_ + dh) * S_ + s0;
        const int stride = (MODE == 0) ? DH_ : 1;
#pragma unroll
        for (int mi = 0; mi < 2; ++mi)
#pragma unroll
            for (int g = 0; g < 8; ++g)
                po[(size_t)(mi * 16 + g) * stride] =
                    f32_to_bf16(acc[mi * 4 + t][g] + bv);
    }
}

// ---------------------------------------------------------------------------
// Kernel 3: logits = (Qh @ Khᵀ) * 1/sqrt(DH), fused multiplicative mask.
// One wave owns (b,h, 32-row strip): 4 resident Q-frags, sweeps 128 key
// tiles; each key tile = 2 B-frag loads feeding 4 WMMAs.
// Mask loads and attn stores walk running pointers with immediate offsets;
// attn stores are non-temporal (1GB stream, re-read only by softmax).
// ---------------------------------------------------------------------------
__global__ __launch_bounds__(128) void attn_logits(
    const unsigned short* __restrict__ qp, const unsigned short* __restrict__ kp,
    const float* __restrict__ mask, float* __restrict__ attn) {
    const int wave = blockIdx.x * 4 + (threadIdx.x >> 5);  // 4096 waves
    const int bh = wave >> 6;          // 64
    const int mt = wave & 63;          // 64 row blocks of 32
    const int b = bh >> 4;
    const int lane = threadIdx.x & 31;
    const int half = lane >> 4;
    const int r = lane & 15;

    const unsigned short* __restrict__ qrow0 =
        qp + ((size_t)bh * S_ + mt * 32 + r) * DH_ + (half << 3);
    const unsigned short* __restrict__ qrow1 = qrow0 + 16 * DH_;
    Frag qa00 = load_a_bf16(qrow0);        // rows 0..15, K 0..31
    Frag qa01 = load_a_bf16(qrow0 + 32);   // rows 0..15, K 32..63
    Frag qa10 = load_a_bf16(qrow1);        // rows 16..31
    Frag qa11 = load_a_bf16(qrow1 + 32);

    const unsigned short* __restrict__ pk =
        kp + ((size_t)bh * S_ + r) * DH_ + (half << 3);
    const float* __restrict__ pm =
        mask + ((size_t)b * S_ + mt * 32 + (half << 3)) * S_ + r;
    float* __restrict__ pa =
        attn + ((size_t)bh * S_ + mt * 32 + (half << 3)) * S_ + r;

    for (int nt = 0; nt < 128; ++nt) {
        Frag kb0 = load_b_bf16(pk);
        Frag kb1 = load_b_bf16(pk + 32);

        v8f acc0 = {}, acc1 = {};
        acc0 = wmma_bf16(qa00, kb0, acc0);
        acc0 = wmma_bf16(qa01, kb1, acc0);
        acc1 = wmma_bf16(qa10, kb0, acc1);
        acc1 = wmma_bf16(qa11, kb1, acc1);

#pragma unroll
        for (int mi = 0; mi < 2; ++mi) {
            const v8f& acc = (mi == 0) ? acc0 : acc1;
#pragma unroll
            for (int g = 0; g < 8; ++g) {
                const int off = (mi * 16 + g) * S_;   // compile-time
                float mk = pm[off];
                // v*mk - 1e6*(1-mk)  ==  (v + 1e6)*mk - 1e6
                float v = fmaf(acc[g], 0.125f, 1e6f) * mk - 1e6f;
                __builtin_nontemporal_store(v, pa + off);
            }
        }
        pk += 16 * DH_;   // next 16 key rows
        pm += 16;
        pa += 16;
    }
}

// ---------------------------------------------------------------------------
// Kernel 4: in-place row softmax over 2048 columns. One 256-thread block/row.
// ---------------------------------------------------------------------------
__global__ __launch_bounds__(256) void softmax_rows(float* __restrict__ attn) {
    float* __restrict__ p = attn + (size_t)blockIdx.x * S_;
    __shared__ float red[256];
    const int t = threadIdx.x;

    float vals[8];
    float mx = -3.4e38f;
#pragma unroll
    for (int i = 0; i < 8; ++i) {
        vals[i] = p[t + i * 256];
        mx = fmaxf(mx, vals[i]);
    }
    red[t] = mx; __syncthreads();
    for (int s = 128; s > 0; s >>= 1) {
        if (t < s) red[t] = fmaxf(red[t], red[t + s]);
        __syncthreads();
    }
    mx = red[0]; __syncthreads();

    float sum = 0.0f;
#pragma unroll
    for (int i = 0; i < 8; ++i) {
        vals[i] = __expf(vals[i] - mx);
        sum += vals[i];
    }
    red[t] = sum; __syncthreads();
    for (int s = 128; s > 0; s >>= 1) {
        if (t < s) red[t] += red[t + s];
        __syncthreads();
    }
    const float inv = 1.0f / red[0];
#pragma unroll
    for (int i = 0; i < 8; ++i) p[t + i * 256] = vals[i] * inv;
}

// ---------------------------------------------------------------------------
// Kernel 5: ctx = attn(f32 -> bf16 on load, non-temporal) @ Vhᵀ-stored.
// One wave: 32 rows x full DH (64) -> 8 WMMAs per k-step, 64 k-steps.
// Stores ctx bf16 in concat layout [B, S, D].
// ---------------------------------------------------------------------------
__global__ __launch_bounds__(128) void ctx_gemm(
    const float* __restrict__ attn, const unsigned short* __restrict__ vT,
    unsigned short* __restrict__ ctx) {
    const int wave = blockIdx.x * 4 + (threadIdx.x >> 5);  // 4096 waves
    const int bh = wave >> 6;          // 64
    const int mt = wave & 63;          // 64 row blocks of 32
    const int b = bh >> 4, h = bh & 15;
    const int lane = threadIdx.x & 31;
    const int half = lane >> 4;
    const int r = lane & 15;

    const float* __restrict__ pa0 =
        attn + ((size_t)bh * S_ + mt * 32 + r) * S_ + (half << 3);
    const float* __restrict__ pa1 = pa0 + 16 * S_;
    const unsigned short* __restrict__ pb[4];
#pragma unroll
    for (int t = 0; t < 4; ++t)
        pb[t] = vT + ((size_t)bh * DH_ + t * 16 + r) * S_ + (half << 3);

    v8f acc[8] = {};                   // acc[mi*4 + t]
    for (int k0 = 0; k0 < S_; k0 += 32) {
        __builtin_prefetch((const void*)(pa0 + 128), 0, 1);
        __builtin_prefetch((const void*)(pb[0] + 128), 0, 1);
        Frag fa0 = load_a_f32<true>(pa0);
        Frag fa1 = load_a_f32<true>(pa1);
        Frag fb[4];
#pragma unroll
        for (int t = 0; t < 4; ++t) fb[t] = load_b_bf16(pb[t]);
#pragma unroll
        for (int t = 0; t < 4; ++t) {
            acc[t]     = wmma_bf16(fa0, fb[t], acc[t]);
            acc[4 + t] = wmma_bf16(fa1, fb[t], acc[4 + t]);
        }
        pa0 += 32; pa1 += 32;
#pragma unroll
        for (int t = 0; t < 4; ++t) pb[t] += 32;
    }

    unsigned short* __restrict__ pc =
        ctx + ((size_t)b * S_ + mt * 32 + (half << 3)) * D_ + h * DH_ + r;
#pragma unroll
    for (int t = 0; t < 4; ++t)
#pragma unroll
        for (int mi = 0; mi < 2; ++mi)
#pragma unroll
            for (int g = 0; g < 8; ++g)
                pc[(size_t)(mi * 16 + g) * D_ + t * 16] =
                    f32_to_bf16(acc[mi * 4 + t][g]);
}

// ---------------------------------------------------------------------------
// Kernel 6: out = ctx(bf16) @ WoT(bf16) + wo_b, f32 store to d_out.
// One wave computes a 32x64 block.
// ---------------------------------------------------------------------------
__global__ __launch_bounds__(128) void out_gemm(
    const unsigned short* __restrict__ ctx, const unsigned short* __restrict__ WoT,
    const float* __restrict__ bias, float* __restrict__ out) {
    const int wave = blockIdx.x * 4 + (threadIdx.x >> 5);  // 4096 waves
    const int mt = wave >> 4;          // 256 row blocks of 32
    const int ns = wave & 15;          // 16 col strips of 64
    const int lane = threadIdx.x & 31;
    const int half = lane >> 4;
    const int r = lane & 15;

    const unsigned short* __restrict__ pa0 =
        ctx + (size_t)(mt * 32 + r) * 1024 + (half << 3);
    const unsigned short* __restrict__ pa1 = pa0 + 16 * 1024;
    const unsigned short* __restrict__ pb[4];
#pragma unroll
    for (int t = 0; t < 4; ++t)
        pb[t] = WoT + (size_t)(ns * 64 + t * 16 + r) * 1024 + (half << 3);

    v8f acc[8] = {};                   // acc[mi*4 + t]
    for (int k0 = 0; k0 < 1024; k0 += 32) {
        __builtin_prefetch((const void*)(pa0 + 128), 0, 1);
        __builtin_prefetch((const void*)(pb[0] + 128), 0, 1);
        Frag fa0 = load_a_bf16(pa0);
        Frag fa1 = load_a_bf16(pa1);
        Frag fb[4];
#pragma unroll
        for (int t = 0; t < 4; ++t) fb[t] = load_b_bf16(pb[t]);
#pragma unroll
        for (int t = 0; t < 4; ++t) {
            acc[t]     = wmma_bf16(fa0, fb[t], acc[t]);
            acc[4 + t] = wmma_bf16(fa1, fb[t], acc[4 + t]);
        }
        pa0 += 32; pa1 += 32;
#pragma unroll
        for (int t = 0; t < 4; ++t) pb[t] += 32;
    }

    float* __restrict__ po = out + (size_t)(mt * 32 + (half << 3)) * 1024;
#pragma unroll
    for (int t = 0; t < 4; ++t) {
        const int col = ns * 64 + t * 16 + r;
        const float bv = bias[col];
#pragma unroll
        for (int mi = 0; mi < 2; ++mi)
#pragma unroll
            for (int g = 0; g < 8; ++g)
                po[(size_t)(mi * 16 + g) * 1024 + col] = acc[mi * 4 + t][g] + bv;
    }
}

// ---------------------------------------------------------------------------
// Host launcher
// ---------------------------------------------------------------------------
extern "C" void kernel_launch(void* const* d_in, const int* in_sizes, int n_in,
                              void* d_out, int out_size, void* d_ws, size_t ws_size,
                              hipStream_t stream) {
    // setup_inputs order: v,k,q,mask, wq_w,wq_b, wk_w,wk_b, wv_w,wv_b, wo_w,wo_b
    const float* v    = (const float*)d_in[0];
    const float* k    = (const float*)d_in[1];
    const float* q    = (const float*)d_in[2];
    const float* mask = (const float*)d_in[3];
    const float* wq_w = (const float*)d_in[4];
    const float* wq_b = (const float*)d_in[5];
    const float* wk_w = (const float*)d_in[6];
    const float* wk_b = (const float*)d_in[7];
    const float* wv_w = (const float*)d_in[8];
    const float* wv_b = (const float*)d_in[9];
    const float* wo_w = (const float*)d_in[10];
    const float* wo_b = (const float*)d_in[11];

    float* out  = (float*)d_out;                 // [B,S,D]
    float* attn = (float*)d_out + OUT_ELEMS;     // [B,H,S,S]

    // Workspace layout (bytes)
    char* ws = (char*)d_ws;
    unsigned short* wqT   = (unsigned short*)(ws + 0);            // 2 MB
    unsigned short* wkT   = (unsigned short*)(ws + (2u << 20));
    unsigned short* wvT   = (unsigned short*)(ws + (4u << 20));
    unsigned short* woT   = (unsigned short*)(ws + (6u << 20));
    unsigned short* qproj = (unsigned short*)(ws + (8u << 20));   // 16 MB [B,H,S,DH]
    unsigned short* kproj = (unsigned short*)(ws + (24u << 20));  // 16 MB [B,H,S,DH]
    unsigned short* vprojT= (unsigned short*)(ws + (40u << 20));  // 16 MB [B,H,DH,S]
    unsigned short* ctx   = (unsigned short*)(ws + (56u << 20));  // 16 MB [B,S,D]

    // 1) weight transpose + bf16 convert
    transpose_w_bf16<<<4096, 256, 0, stream>>>(wq_w, wqT);
    transpose_w_bf16<<<4096, 256, 0, stream>>>(wk_w, wkT);
    transpose_w_bf16<<<4096, 256, 0, stream>>>(wv_w, wvT);
    transpose_w_bf16<<<4096, 256, 0, stream>>>(wo_w, woT);

    // 2) Q/K/V projections (4096 waves each, 4 waves/block)
    proj_gemm<0><<<1024, 128, 0, stream>>>(q, wqT, wq_b, qproj);
    proj_gemm<0><<<1024, 128, 0, stream>>>(k, wkT, wk_b, kproj);
    proj_gemm<1><<<1024, 128, 0, stream>>>(v, wvT, wv_b, vprojT);

    // 3) scaled+masked logits straight into the attn slice of d_out
    attn_logits<<<1024, 128, 0, stream>>>(qproj, kproj, mask, attn);

    // 4) in-place softmax over each of B*H*S rows
    softmax_rows<<<B_ * H_ * S_, 256, 0, stream>>>(attn);

    // 5) ctx = attn @ Vh
    ctx_gemm<<<1024, 128, 0, stream>>>(attn, vprojT, ctx);

    // 6) output projection
    out_gemm<<<1024, 128, 0, stream>>>(ctx, woT, wo_b, out);
}
